// ProposalBranch_loc_41626823033501
// MI455X (gfx1250) — matrix operand order repaired and checked by reading, not compiled, not run on hardware
//
#include <hip/hip_runtime.h>
#include <hip/hip_bf16.h>

typedef __bf16 bf16t;
typedef __attribute__((ext_vector_type(8)))  bf16t v8bf;
typedef __attribute__((ext_vector_type(16))) bf16t v16bf;
typedef __attribute__((ext_vector_type(8)))  float v8f;

#define BB   4
#define TN   2048   // N columns for every GEMM (and T of feature)
#define TFR  4096   // frame-level T

__device__ __forceinline__ int clampi(int v, int lo, int hi) {
    return v < lo ? lo : (v > hi ? hi : v);
}

// ---------------------------------------------------------------------------
// bf16 WMMA GEMM:  Y[b][m][n] = sum_k W[m][k] * X[b][k][n] + bias[m]
// block = 256 thr (8 waves), tile 128(M) x 128(N), K-step 32, LDS staged.
// wave (wm,wn) owns a 32x64 subtile = 2x4 wmma accumulators.
// ---------------------------------------------------------------------------
__global__ __launch_bounds__(256) void gemm_bf16_wmma(
    float* __restrict__ Y, const bf16t* __restrict__ X,
    const bf16t* __restrict__ W, const float* __restrict__ bias,
    int M, int K)
{
    __shared__ __align__(32) bf16t Al[128 * 32];   // [m][k]
    __shared__ __align__(32) bf16t Bl[32 * 128];   // [k][n]

    const int b    = blockIdx.z;
    const int n0   = blockIdx.x * 128;
    const int m0   = blockIdx.y * 128;
    const int tid  = threadIdx.x;
    const int lane = tid & 31;
    const int wave = tid >> 5;
    const int wm   = wave & 3;    // 4 waves along M
    const int wn   = wave >> 2;   // 2 waves along N
    const bf16t* Xb = X + (size_t)b * K * TN;

    v8f acc[2][4];
    const v8f vzero = {0.f, 0.f, 0.f, 0.f, 0.f, 0.f, 0.f, 0.f};
#pragma unroll
    for (int i = 0; i < 2; ++i)
#pragma unroll
        for (int j = 0; j < 4; ++j) acc[i][j] = vzero;

    const int mlane = lane & 15;
    const int koff  = (lane & 16) ? 8 : 0;

    for (int k0 = 0; k0 < K; k0 += 32) {
        __syncthreads();
        // A tile 128x32: 512 chunks of 8 bf16 (16B) -> 2 passes of 256 threads
#pragma unroll
        for (int p = 0; p < 2; ++p) {
            int c   = p * 256 + tid;
            int row = c >> 2;
            int col = (c & 3) * 8;
            *(v8bf*)&Al[row * 32 + col] =
                *(const v8bf*)&W[(size_t)(m0 + row) * K + k0 + col];
        }
        // B tile 32x128
#pragma unroll
        for (int p = 0; p < 2; ++p) {
            int c   = p * 256 + tid;
            int row = c >> 4;
            int col = (c & 15) * 8;
            *(v8bf*)&Bl[row * 128 + col] =
                *(const v8bf*)&Xb[(size_t)(k0 + row) * TN + n0 + col];
        }
        __syncthreads();

        v16bf afrag[2], bfrag[4];
#pragma unroll
        for (int ms = 0; ms < 2; ++ms) {
            const bf16t* ap = &Al[(wm * 32 + ms * 16 + mlane) * 32];
            union { v16bf v; v8bf h[2]; } u;
            u.h[0] = *(const v8bf*)(ap + koff);        // K = koff .. koff+7
            u.h[1] = *(const v8bf*)(ap + koff + 16);   // K = koff+16 .. koff+23
            afrag[ms] = u.v;
        }
#pragma unroll
        for (int ns = 0; ns < 4; ++ns)   // lane l holds column N=l%16, K in table order (koff..koff+7, koff+16..koff+23)
#pragma unroll
            for (int s = 0; s < 16; ++s)
                bfrag[ns][s] = Bl[(koff + s + (s >= 8 ? 8 : 0)) * 128 + wn * 64 + ns * 16 + mlane];

#pragma unroll
        for (int ms = 0; ms < 2; ++ms)
#pragma unroll
            for (int ns = 0; ns < 4; ++ns)
                acc[ms][ns] = __builtin_amdgcn_wmma_f32_16x16x32_bf16(
                    false, afrag[ms], false, bfrag[ns],
                    (short)0, acc[ms][ns], false, false);
    }

    const int mro  = (lane & 16) ? 8 : 0;
    const int ncol = lane & 15;
#pragma unroll
    for (int ms = 0; ms < 2; ++ms)
#pragma unroll
        for (int ns = 0; ns < 4; ++ns)
#pragma unroll
            for (int r = 0; r < 8; ++r) {
                int m = m0 + wm * 32 + ms * 16 + r + mro;
                int n = n0 + wn * 64 + ns * 16 + ncol;
                Y[((size_t)b * M + m) * TN + n] = acc[ms][ns][r] + bias[m];
            }
}

// ---------------------------------------------------------------------------
// GroupNorm stats: one block per (b, group); group = contiguous cpg*TN floats
// ---------------------------------------------------------------------------
__global__ __launch_bounds__(256) void gn_stats(
    const float* __restrict__ Y, float* __restrict__ stats, int M)
{
    const int bg = blockIdx.x;
    const int b = bg >> 5, g = bg & 31;
    const int cpg   = M >> 5;
    const int count = cpg * TN;
    const float* base = Y + ((size_t)b * M + (size_t)g * cpg) * TN;
    float s = 0.f, ss = 0.f;
    for (int i = threadIdx.x; i < count; i += 256) {
        float v = base[i]; s += v; ss += v * v;
    }
    __shared__ float rs[256], rq[256];
    rs[threadIdx.x] = s; rq[threadIdx.x] = ss;
    __syncthreads();
    for (int off = 128; off > 0; off >>= 1) {
        if (threadIdx.x < off) {
            rs[threadIdx.x] += rs[threadIdx.x + off];
            rq[threadIdx.x] += rq[threadIdx.x + off];
        }
        __syncthreads();
    }
    if (threadIdx.x == 0) {
        float mean = rs[0] / (float)count;
        float var  = rq[0] / (float)count - mean * mean;
        stats[bg * 2 + 0] = mean;
        stats[bg * 2 + 1] = rsqrtf(var + 1e-5f);
    }
}

// ---------------------------------------------------------------------------
// GroupNorm apply + ReLU; optional f32 output and/or bf16 write into cat[]
// ---------------------------------------------------------------------------
__global__ __launch_bounds__(256) void gn_apply(
    const float* __restrict__ Y, const float* __restrict__ stats,
    const float* __restrict__ gamma, const float* __restrict__ beta,
    int M, float* __restrict__ outf, bf16t* __restrict__ outb, int choff)
{
    size_t idx   = (size_t)blockIdx.x * 256 + threadIdx.x;
    size_t total = (size_t)BB * M * TN;
    if (idx >= total) return;
    int n = (int)(idx & (TN - 1));
    int m = (int)((idx >> 11) % M);
    int b = (int)(idx / ((size_t)M * TN));
    int g = m / (M >> 5);
    float mean = stats[(b * 32 + g) * 2];
    float rstd = stats[(b * 32 + g) * 2 + 1];
    float v = (Y[idx] - mean) * rstd * gamma[m] + beta[m];
    v = fmaxf(v, 0.f);
    if (outf) outf[idx] = v;
    if (outb) outb[((size_t)b * 1536 + choff + m) * TN + n] = (bf16t)v;
}

// ---------------------------------------------------------------------------
// 32-wide chunk maxima for range-max queries
// ---------------------------------------------------------------------------
__global__ __launch_bounds__(256) void chunkmax_k(
    const float* __restrict__ x, float* __restrict__ cm, int C, int Tt)
{
    int nj = Tt >> 5;
    size_t idx   = (size_t)blockIdx.x * 256 + threadIdx.x;
    size_t total = (size_t)BB * C * nj;
    if (idx >= total) return;
    int j = (int)(idx % nj);
    size_t bc = idx / nj;
    const float* p = x + bc * Tt + (size_t)j * 32;
    float m = p[0];
#pragma unroll
    for (int i = 1; i < 32; ++i) m = fmaxf(m, p[i]);
    cm[idx] = m;
}

__device__ __forceinline__ float range_max(
    const float* __restrict__ x, const float* __restrict__ cm, int s, int e)
{
    int cs = s >> 5, ce = e >> 5;
    float m = -3.4e38f;
    if (cs == ce) {
        for (int t = s; t <= e; ++t) m = fmaxf(m, x[t]);
        return m;
    }
    int b1 = (cs + 1) << 5;
    for (int t = s; t < b1; ++t)       m = fmaxf(m, x[t]);
    for (int j = cs + 1; j < ce; ++j)  m = fmaxf(m, cm[j]);
    for (int t = ce << 5; t <= e; ++t) m = fmaxf(m, x[t]);
    return m;
}

// boundary max pool over feat [B,512,T] -> cat channels 256..767 (bf16)
__global__ __launch_bounds__(256) void pool_feat_k(
    const float* __restrict__ feat, const float* __restrict__ cm,
    const float* __restrict__ seg, bf16t* __restrict__ cat)
{
    int n = blockIdx.x, b = blockIdx.y, t = threadIdx.x;
    const float* sg = seg + ((size_t)b * TN + n) * 4;
    const int T = TN;
    int sL = clampi((int)floorf(sg[0]), 0, T - 1);
    int eL = clampi((int)ceilf (sg[1]), 0, T - 1); eL = eL > sL ? eL : sL;
    int sR = clampi((int)floorf(sg[2]), 0, T - 1);
    int eR = clampi((int)ceilf (sg[3]), 0, T - 1); eR = eR > sR ? eR : sR;
    {
        size_t c = (size_t)b * 512 + t;           // left half channels 0..255
        float v = range_max(feat + c * T, cm + c * 64, sL, eL);
        cat[((size_t)b * 1536 + 256 + t) * TN + n] = (bf16t)v;
    }
    {
        size_t c = (size_t)b * 512 + 256 + t;     // right half channels 256..511
        float v = range_max(feat + c * T, cm + c * 64, sR, eR);
        cat[((size_t)b * 1536 + 512 + t) * TN + n] = (bf16t)v;
    }
}

// boundary max pool over frame_level_feature [B,256,Tf] -> roi_bf [B,256,N]
__global__ __launch_bounds__(256) void pool_roi_k(
    const float* __restrict__ fr, const float* __restrict__ cm,
    const float* __restrict__ seg, bf16t* __restrict__ roib)
{
    int n = blockIdx.x, b = blockIdx.y, t = threadIdx.x;
    const float* sg = seg + ((size_t)b * TN + n) * 4;
    const int T = TFR;
    int s, e;
    if (t < 128) {
        s = clampi((int)floorf(sg[0]), 0, T - 1);
        e = clampi((int)ceilf (sg[1]), 0, T - 1); e = e > s ? e : s;
    } else {
        s = clampi((int)floorf(sg[2]), 0, T - 1);
        e = clampi((int)ceilf (sg[3]), 0, T - 1); e = e > s ? e : s;
    }
    size_t c = (size_t)b * 256 + t;
    float v = range_max(fr + c * T, cm + c * 128, s, e);
    roib[((size_t)b * 256 + t) * TN + n] = (bf16t)v;
}

// ---------------------------------------------------------------------------
// conversions / staging
// ---------------------------------------------------------------------------
__global__ void cvt_bf16_k(const float* __restrict__ s, bf16t* __restrict__ d, size_t n)
{
    size_t i = (size_t)blockIdx.x * 256 + threadIdx.x;
    if (i < n) d[i] = (bf16t)s[i];
}

__global__ void cvt_wprop_k(const float* __restrict__ s, bf16t* __restrict__ d)
{   // [512][1524] f32 -> [512][1536] bf16, zero-padded K
    size_t i = (size_t)blockIdx.x * 256 + threadIdx.x;
    if (i >= (size_t)512 * 1536) return;
    int k = (int)(i % 1536); int m = (int)(i / 1536);
    d[i] = (k < 1524) ? (bf16t)s[(size_t)m * 1524 + k] : (bf16t)0.f;
}

__global__ void cvt_cat_k(const float* __restrict__ s, bf16t* __restrict__ cat,
                          int C, int choff)
{   // [B][C][TN] f32 -> cat[b][choff+c][n]
    size_t i = (size_t)blockIdx.x * 256 + threadIdx.x;
    if (i >= (size_t)BB * C * TN) return;
    int n = (int)(i & (TN - 1));
    int c = (int)((i >> 11) % C);
    int b = (int)(i / ((size_t)C * TN));
    cat[((size_t)b * 1536 + choff + c) * TN + n] = (bf16t)s[i];
}

__global__ void zero_pad_k(bf16t* __restrict__ cat)
{   // cat rows 1524..1535 = 0
    size_t i = (size_t)blockIdx.x * 256 + threadIdx.x;
    if (i >= (size_t)BB * 12 * TN) return;
    int n = (int)(i & (TN - 1));
    int c = (int)((i >> 11) % 12);
    int b = (int)(i / ((size_t)12 * TN));
    cat[((size_t)b * 1536 + 1524 + c) * TN + n] = (bf16t)0.f;
}

// ---------------------------------------------------------------------------
extern "C" void kernel_launch(void* const* d_in, const int* in_sizes, int n_in,
                              void* d_out, int out_size, void* d_ws, size_t ws_size,
                              hipStream_t stream)
{
    const float* feature = (const float*)d_in[0];
    const float* frame   = (const float*)d_in[1];
    const float* segs    = (const float*)d_in[2];
    const float* fsegs   = (const float*)d_in[3];
    const float* lastcls = (const float*)d_in[4];
    const float* confft  = (const float*)d_in[5];
    const float* w_cur  = (const float*)d_in[6];  const float* b_cur  = (const float*)d_in[7];
    const float* g_cur  = (const float*)d_in[8];  const float* be_cur = (const float*)d_in[9];
    const float* w_lr   = (const float*)d_in[10]; const float* b_lr   = (const float*)d_in[11];
    const float* g_lr   = (const float*)d_in[12]; const float* be_lr  = (const float*)d_in[13];
    const float* w_roi  = (const float*)d_in[14]; const float* b_roi  = (const float*)d_in[15];
    const float* g_roi  = (const float*)d_in[16]; const float* be_roi = (const float*)d_in[17];
    const float* w_prop = (const float*)d_in[18]; const float* b_prop = (const float*)d_in[19];
    const float* g_prop = (const float*)d_in[20]; const float* be_prop= (const float*)d_in[21];
    (void)in_sizes; (void)n_in; (void)out_size; (void)ws_size;

    float* out  = (float*)d_out;                       // [4,512,2048]
    float* feat = out + (size_t)BB * 512 * TN;         // [4,512,2048] (tuple item 2)

    char* ws = (char*)d_ws;
    size_t off = 0;
    auto alloc = [&](size_t bytes) {
        char* p = ws + off;
        off = (off + bytes + 255) & ~(size_t)255;
        return (void*)p;
    };
    bf16t* xbf   = (bf16t*)alloc((size_t)BB * 512 * TN * 2);   // feature bf16
    bf16t* wcur  = (bf16t*)alloc((size_t)256 * 512 * 2);
    bf16t* wlr   = (bf16t*)alloc((size_t)512 * 512 * 2);
    bf16t* wroi  = (bf16t*)alloc((size_t)256 * 256 * 2);
    bf16t* wprop = (bf16t*)alloc((size_t)512 * 1536 * 2);
    float* ybuf  = (float*)alloc((size_t)BB * 512 * TN * 4);   // reused conv scratch
    bf16t* cat   = (bf16t*)alloc((size_t)BB * 1536 * TN * 2);
    bf16t* roib  = (bf16t*)alloc((size_t)BB * 256 * TN * 2);
    float* cmf   = (float*)alloc((size_t)BB * 512 * 64 * 4);
    float* cmr   = (float*)alloc((size_t)BB * 256 * 128 * 4);
    float* stats = (float*)alloc((size_t)BB * 32 * 2 * 4);

    auto nblk = [](size_t n) { return (unsigned)((n + 255) / 256); };

    // stage 0: conversions
    cvt_bf16_k<<<nblk((size_t)BB * 512 * TN), 256, 0, stream>>>(feature, xbf, (size_t)BB * 512 * TN);
    cvt_bf16_k<<<nblk((size_t)256 * 512), 256, 0, stream>>>(w_cur, wcur, (size_t)256 * 512);
    cvt_bf16_k<<<nblk((size_t)512 * 512), 256, 0, stream>>>(w_lr, wlr, (size_t)512 * 512);
    cvt_bf16_k<<<nblk((size_t)256 * 256), 256, 0, stream>>>(w_roi, wroi, (size_t)256 * 256);
    cvt_wprop_k<<<nblk((size_t)512 * 1536), 256, 0, stream>>>(w_prop, wprop);

    // stage 1: fm_short = block(feature, w_cur)  -> cat[768:1024] (bf16)
    gemm_bf16_wmma<<<dim3(16, 2, BB), 256, 0, stream>>>(ybuf, xbf, wcur, b_cur, 256, 512);
    gn_stats<<<BB * 32, 256, 0, stream>>>(ybuf, stats, 256);
    gn_apply<<<nblk((size_t)BB * 256 * TN), 256, 0, stream>>>(ybuf, stats, g_cur, be_cur, 256, nullptr, cat, 768);

    // stage 2: feat = block(feature, w_lr)  -> d_out second half (f32)
    gemm_bf16_wmma<<<dim3(16, 4, BB), 256, 0, stream>>>(ybuf, xbf, wlr, b_lr, 512, 512);
    gn_stats<<<BB * 32, 256, 0, stream>>>(ybuf, stats, 512);
    gn_apply<<<nblk((size_t)BB * 512 * TN), 256, 0, stream>>>(ybuf, stats, g_lr, be_lr, 512, feat, nullptr, 0);

    // boundary max pooling
    chunkmax_k<<<nblk((size_t)BB * 512 * 64), 256, 0, stream>>>(feat, cmf, 512, TN);
    pool_feat_k<<<dim3(TN, BB), 256, 0, stream>>>(feat, cmf, segs, cat);
    chunkmax_k<<<nblk((size_t)BB * 256 * 128), 256, 0, stream>>>(frame, cmr, 256, TFR);
    pool_roi_k<<<dim3(TN, BB), 256, 0, stream>>>(frame, cmr, fsegs, roib);

    // stage 3: prop_roi = block(prop_roi, w_roi) -> cat[0:256]
    gemm_bf16_wmma<<<dim3(16, 2, BB), 256, 0, stream>>>(ybuf, roib, wroi, b_roi, 256, 256);
    gn_stats<<<BB * 32, 256, 0, stream>>>(ybuf, stats, 256);
    gn_apply<<<nblk((size_t)BB * 256 * TN), 256, 0, stream>>>(ybuf, stats, g_roi, be_roi, 256, nullptr, cat, 0);

    // cat extras: lastclass [1024:1274], Conf [1274:1524], zero pad [1524:1536]
    cvt_cat_k<<<nblk((size_t)BB * 250 * TN), 256, 0, stream>>>(lastcls, cat, 250, 1024);
    cvt_cat_k<<<nblk((size_t)BB * 250 * TN), 256, 0, stream>>>(confft, cat, 250, 1274);
    zero_pad_k<<<nblk((size_t)BB * 12 * TN), 256, 0, stream>>>(cat);

    // stage 4: out = block(cat, w_prop) -> d_out first half (f32)
    gemm_bf16_wmma<<<dim3(16, 4, BB), 256, 0, stream>>>(ybuf, cat, wprop, b_prop, 512, 1536);
    gn_stats<<<BB * 32, 256, 0, stream>>>(ybuf, stats, 512);
    gn_apply<<<nblk((size_t)BB * 512 * TN), 256, 0, stream>>>(ybuf, stats, g_prop, be_prop, 512, out, nullptr, 0);
}
